// NonLoacalInteraction_16844861735176
// MI455X (gfx1250) — matrix operand order repaired and checked by reading, not compiled, not run on hardware
//
#include <hip/hip_runtime.h>
#include <hip/hip_bf16.h>
#include <math.h>

// ---------------------------------------------------------------------------
// Multi-head self-attention (flash-attention style) for gfx1250 / MI455X.
//   N_A=4096 tokens, N_F=256 features, NH=8 heads, D=32 head dim.
//   Pipeline:
//     1) f32->f16 conversion of x and Wq/Wk/Wv (once, to workspace)
//     2) q/k/v = xh @ Wh^T + b via v_wmma_f32_16x16x32_f16 (f16 in ws)
//     3) flash attention, double-buffered: K tiles streamed by the Tensor
//        Data Mover (tensor_load_to_lds / TENSORcnt), V tiles register-
//        prefetched one iteration ahead and transposed into LDS, online
//        softmax, both gemms on WMMA with f32 accumulation.
// ---------------------------------------------------------------------------

typedef _Float16 v16h __attribute__((ext_vector_type(16)));
typedef _Float16 v8h  __attribute__((ext_vector_type(8)));
typedef _Float16 v4h  __attribute__((ext_vector_type(4)));
typedef float    v8f  __attribute__((ext_vector_type(8)));
typedef float    v4f  __attribute__((ext_vector_type(4)));
typedef unsigned int  u32x4 __attribute__((ext_vector_type(4)));
typedef int           i32x8 __attribute__((ext_vector_type(8)));
typedef int           i32x4 __attribute__((ext_vector_type(4)));

#define N_A   4096
#define N_F   256
#define NH    8
#define DH    32
#define QTILE 128   // query rows per block (8 waves x 16 rows)
#define KTILE 64    // keys staged in LDS per iteration
#define NWAVE 8
#define NT    (N_A / KTILE)

// ---- f16 operand assembler from two 16-byte chunks ------------------------
__device__ __forceinline__ v16h pack16(v8h lo, v8h hi) {
  v16h r;
#pragma unroll
  for (int i = 0; i < 8; ++i) { r[i] = lo[i]; r[i + 8] = hi[i]; }
  return r;
}

// ---- Tensor Data Mover: 2D f16 tile, global -> LDS (D# per ISA ch.8) ------
// 6-arg builtin on this toolchain:
//   (uint32x4 group0, int32x8 group1, int32x4 group2, int32x4 group3,
//    int32x8 extra, i32 cpol)
__device__ __forceinline__ void tdm_load_2d_f16(unsigned lds_off,
                                                const void* gptr,
                                                unsigned tensor_d0,
                                                unsigned tensor_d1,
                                                unsigned tile_d0,
                                                unsigned tile_d1,
                                                unsigned long long stride0) {
  const unsigned long long ga = (unsigned long long)(uintptr_t)gptr;
  u32x4 g0;
  g0[0] = 1u;                                              // count=1, user D#
  g0[1] = lds_off;                                         // lds_addr (bytes)
  g0[2] = (unsigned)(ga & 0xffffffffu);                    // global_addr lo
  g0[3] = (unsigned)((ga >> 32) & 0x01ffffffu) | (2u << 30);   // hi + type=2
  i32x8 g1;
  g1[0] = (int)(1u << 16);                                 // data_size=2B
  g1[1] = (int)((tensor_d0 & 0xffffu) << 16);              // tensor_dim0 lo
  g1[2] = (int)(((tensor_d0 >> 16) & 0xffffu) |
                ((tensor_d1 & 0xffffu) << 16));            // td0 hi | td1 lo
  g1[3] = (int)(((tensor_d1 >> 16) & 0xffffu) |
                ((tile_d0 & 0xffffu) << 16));              // td1 hi | tile_d0
  g1[4] = (int)(tile_d1 & 0xffffu);                        // tile_d1, tile_d2=0
  g1[5] = (int)(unsigned)(stride0 & 0xffffffffull);        // dim0_stride lo
  g1[6] = (int)(unsigned)((stride0 >> 32) & 0xffffull);    // dim0_stride hi
  g1[7] = 0;                                               // dim1_stride=0 (2D)
  const i32x4 z4 = {0, 0, 0, 0};
  const i32x8 z8 = {0, 0, 0, 0, 0, 0, 0, 0};
  __builtin_amdgcn_tensor_load_to_lds(g0, g1, z4, z4, z8, 0);
}

// ===========================================================================
// Kernel 0: fp32 -> f16 bulk conversion
// ===========================================================================
__global__ __launch_bounds__(256)
void f32_to_f16(const float* __restrict__ src, _Float16* __restrict__ dst,
                int n4) {
  const int i = blockIdx.x * blockDim.x + threadIdx.x;
  if (i < n4) {
    const v4f f = reinterpret_cast<const v4f*>(src)[i];
    v4h h;
#pragma unroll
    for (int j = 0; j < 4; ++j) h[j] = (_Float16)f[j];
    reinterpret_cast<v4h*>(dst)[i] = h;
  }
}

// ===========================================================================
// Kernel 1: q/k/v = xh @ Wh^T + b, f16 operands, f32 accumulate, f16 out.
// ===========================================================================
__global__ __launch_bounds__(256)
void qkv_proj(const _Float16* __restrict__ xh,
              const _Float16* __restrict__ Wqh, const float* __restrict__ bq,
              const _Float16* __restrict__ Wkh, const float* __restrict__ bk,
              const _Float16* __restrict__ Wvh, const float* __restrict__ bv,
              _Float16* __restrict__ qh, _Float16* __restrict__ kh,
              _Float16* __restrict__ vh, float qscale)
{
  const _Float16* W; const float* bias; _Float16* out; float scale;
  if (blockIdx.z == 0)      { W = Wqh; bias = bq; out = qh; scale = qscale; }
  else if (blockIdx.z == 1) { W = Wkh; bias = bk; out = kh; scale = 1.0f; }
  else                      { W = Wvh; bias = bv; out = vh; scale = 1.0f; }

  const int wave = threadIdx.x >> 5;
  const int lane = threadIdx.x & 31;
  const int hb   = lane >> 4;
  const int l15  = lane & 15;
  const int m0   = blockIdx.x * QTILE + wave * 16;
  const int j0   = blockIdx.y * 16;

  const int mrow = m0 + l15;        // A row this lane feeds
  const int ncol = j0 + l15;        // B column this lane feeds
  const int cb   = hb ? 8 : 0;      // A-layout: K block {koff..koff+15}
  const int koff = hb ? 16 : 0;     // B-layout: K block {koff..koff+15}

  v8f c = {};
#pragma unroll
  for (int kb = 0; kb < N_F; kb += 32) {
    const _Float16* ap = xh + (size_t)mrow * N_F + kb;
    const v16h a = pack16(*reinterpret_cast<const v8h*>(ap + koff),
                          *reinterpret_cast<const v8h*>(ap + koff + 8));
    const _Float16* bp = W + (size_t)ncol * N_F + kb + koff;   // W^T tile col
    const v16h b = pack16(*reinterpret_cast<const v8h*>(bp),
                          *reinterpret_cast<const v8h*>(bp + 8));
    c = __builtin_amdgcn_wmma_f32_16x16x32_f16(false, a, false, b,
                                               (short)0, c, false, false);
  }

  const float bj = bias[ncol];
#pragma unroll
  for (int j = 0; j < 8; ++j) {
    const int row = m0 + j + hb * 8;               // C layout: M = j + 8*half
    out[(size_t)row * N_F + j0 + l15] = (_Float16)((c[j] + bj) * scale);
  }
}

// ===========================================================================
// Kernel 2: flash attention, double-buffered K (TDM) and V (reg prefetch).
// Block = 8 waves x 16 query rows = 128 rows; one head per blockIdx.y.
// ===========================================================================
__global__ __launch_bounds__(256)
void flash_attn(const _Float16* __restrict__ qh,
                const _Float16* __restrict__ kh,
                const _Float16* __restrict__ vh,
                const unsigned char* __restrict__ mask,
                float* __restrict__ out)
{
  __shared__ _Float16 sK [2][KTILE * DH];         // [buf][key][d]  (TDM target)
  __shared__ _Float16 sVt[2][DH * KTILE];         // [buf][d][key]  transposed
  __shared__ _Float16 sP [NWAVE][16 * KTILE];     // per-wave P staging

  const int h    = blockIdx.y;
  const int q0   = blockIdx.x * QTILE;
  const int tid  = threadIdx.x;
  const int wave = tid >> 5;
  const int lane = tid & 31;
  const int hb   = lane >> 4;
  const int l15  = lane & 15;
  const int cb   = hb ? 8 : 0;
  const int koff = hb ? 16 : 0;

  // --- Q tile for this wave (A operand), resident for whole kernel --------
  const int qrow = q0 + wave * 16 + l15;
  const _Float16* qbase = qh + (size_t)qrow * N_F + h * DH;
  const v16h aQ = pack16(*reinterpret_cast<const v8h*>(qbase + koff),
                         *reinterpret_cast<const v8h*>(qbase + koff + 8));

  // --- online-softmax state: lane<16 owns rows 0..7, lane>=16 rows 8..15 --
  float mrow[8], lrow[8];
  v8f o0 = {}, o1 = {};
#pragma unroll
  for (int j = 0; j < 8; ++j) { mrow[j] = -1e30f; lrow[j] = 0.0f; }

  const int stage_key = tid >> 2;          // 0..63
  const int stage_seg = (tid & 3) * 8;     // 0,8,16,24

  // --- prologue: start tile 0 (K via TDM, V into registers) ---------------
  if (wave == 0) {
    tdm_load_2d_f16((unsigned)(uintptr_t)&sK[0][0],
                    kh + h * DH, DH, KTILE, DH, KTILE, N_F);
  }
  v8h vreg = *reinterpret_cast<const v8h*>(
      vh + (size_t)stage_key * N_F + h * DH + stage_seg);

  for (int kt = 0; kt < NT; ++kt) {
    const int kb  = kt * KTILE;
    const int buf = kt & 1;

    // ---- commit prefetched V(kt) regs -> LDS (transposed) ----------------
#pragma unroll
    for (int i = 0; i < 8; ++i)
      sVt[buf][(stage_seg + i) * KTILE + stage_key] = vreg[i];

    if (wave == 0) __builtin_amdgcn_s_wait_tensorcnt(0);   // TDM(kt) landed
    __syncthreads();   // sK[buf]/sVt[buf] visible; compute(kt-1) finished

    // ---- launch next tile's staging (overlaps with compute below) --------
    if (kt + 1 < NT) {
      if (wave == 0) {
        tdm_load_2d_f16((unsigned)(uintptr_t)&sK[buf ^ 1][0],
                        kh + (size_t)(kb + KTILE) * N_F + h * DH,
                        DH, KTILE, DH, KTILE, N_F);
      }
      vreg = *reinterpret_cast<const v8h*>(
          vh + (size_t)(kb + KTILE + stage_key) * N_F + h * DH + stage_seg);
      if (tid < QTILE)   // warm next mask rows (global_prefetch_b8)
        __builtin_prefetch(mask + (size_t)(q0 + tid) * N_A + kb + KTILE, 0, 1);
    }

    // ---- scores S = (q/TP) @ k^T : four 16x16 tiles, one WMMA each -------
    v8f s[4];
#pragma unroll
    for (int t = 0; t < 4; ++t) {
      const _Float16* kp = &sK[buf][(t * 16 + l15) * DH + koff];
      const v16h bK = pack16(*reinterpret_cast<const v8h*>(kp),
                             *reinterpret_cast<const v8h*>(kp + 8));
      v8f z = {};
      s[t] = __builtin_amdgcn_wmma_f32_16x16x32_f16(false, aQ, false, bK,
                                                    (short)0, z, false, false);
    }

    // ---- mask + online softmax; stage P (f16) into per-wave LDS ----------
#pragma unroll
    for (int j = 0; j < 8; ++j) {
      const int grow = q0 + wave * 16 + j + hb * 8;       // global query row
      const unsigned char* mp = mask + (size_t)grow * N_A + kb + l15;
      float sv[4];
      float rmax = -1e30f;
#pragma unroll
      for (int t = 0; t < 4; ++t) {
        float v = s[t][j];
        if (mp[t * 16]) v = -1e30f;                       // masked -> -inf
        sv[t] = v;
        rmax  = fmaxf(rmax, v);
      }
#pragma unroll
      for (int off = 1; off < 16; off <<= 1)              // reduce 16-lane row
        rmax = fmaxf(rmax, __shfl_xor(rmax, off, 32));
      const float mnew = fmaxf(mrow[j], rmax);
      const float corr = __expf(mrow[j] - mnew);
      float rsum = 0.0f;
#pragma unroll
      for (int t = 0; t < 4; ++t) {
        const float p = __expf(sv[t] - mnew);
        rsum += p;
        sP[wave][(j + hb * 8) * KTILE + t * 16 + l15] = (_Float16)p;
      }
#pragma unroll
      for (int off = 1; off < 16; off <<= 1)
        rsum += __shfl_xor(rsum, off, 32);
      lrow[j] = lrow[j] * corr + rsum;
      mrow[j] = mnew;
      o0[j] *= corr;
      o1[j] *= corr;
    }

    // ---- O += P @ V : K=64 split into two depth-32 WMMAs, 2 col tiles ----
#pragma unroll
    for (int ks = 0; ks < 2; ++ks) {
      const _Float16* pp = &sP[wave][l15 * KTILE + ks * 32];
      const v16h aP = pack16(*reinterpret_cast<const v8h*>(pp + koff),
                             *reinterpret_cast<const v8h*>(pp + koff + 8));
      {
        const _Float16* vp = &sVt[buf][(0 * 16 + l15) * KTILE + ks * 32 + koff];
        const v16h bV = pack16(*reinterpret_cast<const v8h*>(vp),
                               *reinterpret_cast<const v8h*>(vp + 8));
        o0 = __builtin_amdgcn_wmma_f32_16x16x32_f16(false, aP, false, bV,
                                                    (short)0, o0, false, false);
      }
      {
        const _Float16* vp = &sVt[buf][(1 * 16 + l15) * KTILE + ks * 32 + koff];
        const v16h bV = pack16(*reinterpret_cast<const v8h*>(vp),
                               *reinterpret_cast<const v8h*>(vp + 8));
        o1 = __builtin_amdgcn_wmma_f32_16x16x32_f16(false, aP, false, bV,
                                                    (short)0, o1, false, false);
      }
    }
  }

  // ---- finalize: divide by row sums, write fp32 output --------------------
#pragma unroll
  for (int j = 0; j < 8; ++j) {
    const int grow = q0 + wave * 16 + j + hb * 8;
    const float linv = 1.0f / fmaxf(lrow[j], 1e-30f);
    out[(size_t)grow * N_F + h * DH + l15]      = o0[j] * linv;
    out[(size_t)grow * N_F + h * DH + 16 + l15] = o1[j] * linv;
  }
}

// ===========================================================================
// Host launch
// ===========================================================================
extern "C" void kernel_launch(void* const* d_in, const int* in_sizes, int n_in,
                              void* d_out, int out_size, void* d_ws, size_t ws_size,
                              hipStream_t stream) {
  const float*         x  = (const float*)d_in[0];
  const unsigned char* bm = (const unsigned char*)d_in[1];   // jnp bool = u8
  const float*         Wq = (const float*)d_in[2];
  const float*         bq = (const float*)d_in[3];
  const float*         Wk = (const float*)d_in[4];
  const float*         bk = (const float*)d_in[5];
  const float*         Wv = (const float*)d_in[6];
  const float*         bv = (const float*)d_in[7];
  float*               out = (float*)d_out;

  // workspace layout (f16): xh | Wqh | Wkh | Wvh | qh | kh | vh  (~8.4 MB)
  const size_t NX = (size_t)N_A * N_F;    // 1,048,576
  const size_t NW = (size_t)N_F * N_F;    // 65,536
  _Float16* xh  = (_Float16*)d_ws;
  _Float16* Wqh = xh  + NX;
  _Float16* Wkh = Wqh + NW;
  _Float16* Wvh = Wkh + NW;
  _Float16* qh  = Wvh + NW;
  _Float16* kh  = qh  + NX;
  _Float16* vh  = kh  + NX;

  // --- convert x and weights to f16 once ---------------------------------
  f32_to_f16<<<dim3((int)(NX / 4 / 256)), 256, 0, stream>>>(x,  xh,  (int)(NX / 4));
  f32_to_f16<<<dim3((int)(NW / 4 / 256)), 256, 0, stream>>>(Wq, Wqh, (int)(NW / 4));
  f32_to_f16<<<dim3((int)(NW / 4 / 256)), 256, 0, stream>>>(Wk, Wkh, (int)(NW / 4));
  f32_to_f16<<<dim3((int)(NW / 4 / 256)), 256, 0, stream>>>(Wv, Wvh, (int)(NW / 4));

  const float TP = sqrtf(2.0f * (float)DH);   // = 8.0

  dim3 gp(N_A / QTILE, N_F / 16, 3);
  qkv_proj<<<gp, 256, 0, stream>>>(xh, Wqh, bq, Wkh, bk, Wvh, bv,
                                   qh, kh, vh, 1.0f / TP);

  dim3 ga(N_A / QTILE, NH);
  flash_attn<<<ga, 256, 0, stream>>>(qh, kh, vh, bm, out);
}
